// LocalPatternFilter_57595511439939
// MI455X (gfx1250) — matrix operand order repaired and checked by reading, not compiled, not run on hardware
//
#include <hip/hip_runtime.h>

// Problem constants from the reference (B,C,L)=(16,8,131072), K=1024, H=128.
#define NB 16
#define NC 8
#define NL 131072
#define NK 1024
#define NH 128

typedef __attribute__((ext_vector_type(16))) _Float16 v16h;
typedef __attribute__((ext_vector_type(8)))  float    v8f;

union FragU { v16h h; unsigned int u[8]; };
union AccU  { v8f  v; float f[8]; };

// One wave32 per (b,h,c) problem; 4 independent waves (128 threads) per block.
// Circular autocorrelation r[m] is symmetric: r[m] == r[1024-m]. Each wave
// computes lags 0..511 with two WMMA accumulator tiles
//   D_tile[j,i] += A[j,t]*B_tile[t,i],  A[j,t]=w[t0+t-j], B[t,i]=w[t0+t+16i+256*tile]
//   => D_tile[j,i] = r[j + 16*i + 256*tile]
// plus the self-mirrored lag r[512] as a register dot product. The ifftshift
// and the mirror are folded into the output indexing.
__global__ __launch_bounds__(128) void acov_wmma_kernel(const float* __restrict__ X,
                                                        float* __restrict__ out) {
    // Per-wave private LDS regions (no cross-wave sharing -> no barriers).
    __shared__ _Float16     w2[4][2 * NK];   // doubled: w2[p][t] = w[t & 1023]   4x4KB
    __shared__ unsigned int p1[4][NK];       // odd pairs: (w2[2i+1], w2[2i+2])   4x4KB
    __shared__ float        rl[4][520];      // lags 0..512 staging               4x~2KB

    const int tid  = threadIdx.x;
    const int wv   = tid >> 5;
    const int lane = tid & 31;

    const int pid = blockIdx.x * 4 + wv;     // == (b*NH + h)*NC + c
    const int c  = pid % NC;
    const int bh = pid / NC;
    const int h  = bh % NH;
    const int b  = bh / NH;

    // positions = linspace(0, L-K-1-K, H) in f32, truncated to int32
    const float delta = (float)(NL - NK - 1 - NK) / (float)(NH - 1);
    const int   pos   = (int)((float)h * delta);
    const float* xptr = X + (size_t)(b * NC + c) * NL + pos;

    // ---- Phase 1: load window (coalesced per wave), max|x| via butterfly ----
    float xv[32];
    float lmax = 0.0f;
#pragma unroll
    for (int r = 0; r < 32; ++r) {
        float v = xptr[r * 32 + lane];
        xv[r] = v;
        lmax = fmaxf(lmax, fabsf(v));
    }
#pragma unroll
    for (int off = 16; off >= 1; off >>= 1)
        lmax = fmaxf(lmax, __shfl_xor(lmax, off, 32));
    if (lmax == 0.0f) lmax = 1.1920929e-07f;   // eps(float32)
    const float inv = 1.0f / lmax;

    // ---- Phase 2: w = x*hann/fmax; f16 doubled copy in LDS; r[512] in f32 ---
    _Float16* wp = w2[wv];
    float wf[32];
#pragma unroll
    for (int r = 0; r < 32; ++r) {
        int t = r * 32 + lane;
        float win = 0.5f * (1.0f - __cosf(6.283185307179586f * (float)t * (1.0f / (float)NK)));
        float w = xv[r] * win * inv;
        wf[r] = w;
        _Float16 wh = (_Float16)w;
        wp[t]      = wh;
        wp[t + NK] = wh;
    }
    // r[512] = sum_t w[t]*w[(t+512)%1024]: partner element lives in the same
    // lane at register slot (r+16)&31.
    float dot = 0.0f;
#pragma unroll
    for (int r = 0; r < 32; ++r)
        dot = fmaf(wf[r], wf[(r + 16) & 31], dot);
#pragma unroll
    for (int off = 16; off >= 1; off >>= 1)
        dot += __shfl_xor(dot, off, 32);

    // ---- Phase 3: odd-pair array from aligned b32 reads + shifts ------------
    // wu[j] = (w2[2j], w2[2j+1]);  p1[i] = (w2[2i+1], w2[2i+2])
    const unsigned int* wu = (const unsigned int*)wp;
    unsigned int* pp = p1[wv];
#pragma unroll
    for (int r = 0; r < 32; ++r) {
        int i = r * 32 + lane;                      // 0..1023
        unsigned int lo = wu[i];
        unsigned int hi = wu[(i + 1) & (NK - 1)];   // wraps 1024 -> (w[0],w[1])
        pp[i] = (lo >> 16) | (hi << 16);
    }
    // Same-wave LDS RAW: DS ops are in-order per wave; compiler adds dscnt waits.

    // ---- Phase 4: WMMA autocorrelation, lags 0..511 (2 tiles) ---------------
    const int half = lane >> 4;                    // 0: lanes 0-15, 1: lanes 16-31
    const int ncol = lane & 15;                    // A row M == B/D column N
    // A pair parity fixed per lane (ta = 1024 + t0 + ka - ncol; t0,ka even)
    const unsigned int* aBase = (ncol & 1) ? pp : wu;

    AccU acc[2];
#pragma unroll
    for (int tile = 0; tile < 2; ++tile)
        acc[tile].v = (v8f){0.f, 0.f, 0.f, 0.f, 0.f, 0.f, 0.f, 0.f};

#pragma unroll 4
    for (int t0 = 0; t0 < NK; t0 += 32) {
        FragU a;
#pragma unroll
        for (int v = 0; v < 8; ++v) {
            // A 16x32 f16 layout: K = 16*(v>>2) + 8*half + 2*(v&3)
            int ka = 16 * (v >> 2) + 8 * half + 2 * (v & 3);
            a.u[v] = aBase[(NK + t0 + ka - ncol) >> 1];   // affine, no mask
        }
#pragma unroll
        for (int tile = 0; tile < 2; ++tile) {
            FragU bf;
#pragma unroll
            for (int v = 0; v < 8; ++v) {
                // B 32x16 f16 layout: K = 16*(v>>2) + 8*half + 2*(v&3) (always even, aligned b32)
                int tb = t0 + 16 * (v >> 2) + 8 * half + 2 * (v & 3) + 16 * ncol + 256 * tile;
                bf.u[v] = wu[tb >> 1];                    // affine, no mask
            }
            acc[tile].v = __builtin_amdgcn_wmma_f32_16x16x32_f16(
                /*neg_a=*/false, a.h, /*neg_b=*/false, bf.h,
                /*c_mod=*/(short)0, acc[tile].v,
                /*reuse_a=*/false, /*reuse_b=*/false);
        }
    }

    // ---- Phase 5: stage lags 0..512 (D layout: M = v + 8*half, N = ncol) ----
    float* rp = rl[wv];
#pragma unroll
    for (int tile = 0; tile < 2; ++tile)
#pragma unroll
        for (int v = 0; v < 8; ++v) {
            int lag = (v + 8 * half) + 16 * ncol + 256 * tile;   // 0..511
            rp[lag] = acc[tile].f[v];
        }
    if (lane == 0) rp[512] = dot;

    // ---- Phase 6: ifftshift + mirror + coalesced store ----------------------
    // out[k] = r[(k+512) & 1023] = r[min(m, 1024-m)]
    float* optr = out + (size_t)pid * NK;
#pragma unroll
    for (int r = 0; r < 32; ++r) {
        int k = r * 32 + lane;
        int m = (k + (NK / 2)) & (NK - 1);
        int mm = min(m, NK - m);                 // symmetry fold, 0..512
        optr[k] = rp[mm];
    }
}

extern "C" void kernel_launch(void* const* d_in, const int* in_sizes, int n_in,
                              void* d_out, int out_size, void* d_ws, size_t ws_size,
                              hipStream_t stream) {
    (void)in_sizes; (void)n_in; (void)out_size; (void)d_ws; (void)ws_size;
    const float* X = (const float*)d_in[0];   // (16, 8, 131072) f32
    float* out = (float*)d_out;               // (16, 128, 8, 1024) f32

    dim3 grid((NB * NH * NC) / 4);            // 4096 blocks, 4 problems each
    dim3 block(128);                          // 4 independent wave32's
    hipLaunchKernelGGL(acov_wmma_kernel, grid, block, 0, stream, X, out);
}
